// CALDGSeg_59794534695451
// MI455X (gfx1250) — matrix operand order repaired and checked by reading, not compiled, not run on hardware
//
#include <hip/hip_runtime.h>
#include <hip/hip_bf16.h>
#include <math.h>

// ---------------------------------------------------------------------------
// CALDGSeg forward for MI455X (gfx1250, wave32, WMMA + TDM).
// All GEMM-shaped work (convs via implicit im2col, linears) runs through one
// templated WMMA kernel using v_wmma_f32_16x16x32_f16.  Activations are kept
// in NHWC f16 so every WMMA B-fragment lane load is 8 contiguous halfwords.
// For the regular (1x1 / linear) GEMMs the B tile is streamed into LDS by the
// Tensor Data Mover (tensor_load_to_lds + s_wait_tensorcnt) with ping-pong
// double buffering; 3x3 convs use a manual implicit-im2col gather.
// ---------------------------------------------------------------------------

typedef __attribute__((ext_vector_type(16))) _Float16 v16h;
typedef __attribute__((ext_vector_type(8)))  _Float16 v8h;
typedef __attribute__((ext_vector_type(8)))  float    v8f;
typedef __attribute__((ext_vector_type(4)))  unsigned int u32x4;
typedef __attribute__((ext_vector_type(8)))  unsigned int u32x8;

#define SHUF16(lo, hi) __builtin_shufflevector(lo, hi, 0,1,2,3,4,5,6,7,8,9,10,11,12,13,14,15)

static constexpr int   NB    = 2;
static constexpr int   HH    = 112;
static constexpr int   WID   = 112;
static constexpr int   HWSZ  = HH * WID;        // 12544
static constexpr int   NTOK  = NB * HWSZ;       // 25088
static constexpr int   DIMC  = 128;
static constexpr int   HEADS = 8;
static constexpr int   HD    = 16;
static constexpr float EPSV  = 1e-5f;
static constexpr float SCL   = 0.25f;           // HD^-0.5

__device__ __forceinline__ float gelu_exact(float v) {
    return 0.5f * v * (1.0f + erff(v * 0.70710678118654752f));
}

// ---------------------------------------------------------------------------
// TDM: DMA a [64 rows x 32 halfword] tile of a row-major f16 matrix (row
// stride Cbuf elements) into LDS.  pad_enable inserts 4 DWORDs of LDS padding
// after every 16 DWORDs stored, i.e. LDS row stride = 40 halfwords, matching
// the manual staging layout (bank-conflict avoidance for the B-frag reads).
// D# group0/group1 per CDNA5 ISA 8.3/8.4; 2-D tensor -> 2-operand form.
// ---------------------------------------------------------------------------
__device__ __forceinline__ void tdm_load_tile(const _Float16* gptr,
                                              unsigned int lds_off, int Cbuf) {
    unsigned long long ga = (unsigned long long)(uintptr_t)gptr;
    u32x4 g0;
    g0[0] = 1u;                                       // count=1, user mode
    g0[1] = lds_off;                                  // LDS byte address
    g0[2] = (unsigned int)ga;                         // global_addr[31:0]
    g0[3] = (unsigned int)(ga >> 32) | (2u << 30);    // global_addr[56:32] | type=2
    u32x8 g1;
    g1[0] = (1u << 16)        // data_size = 2B
          | (1u << 20)        // pad_enable
          | (3u << 22)        // pad_interval: every 16 DWORDs
          | (3u << 25);       // pad_amount: 4 DWORDs
    g1[1] = ((unsigned int)Cbuf & 0xFFFFu) << 16;     // tensor_dim0[15:0]
    g1[2] = ((unsigned int)NTOK & 0xFFFFu) << 16;     // dim0 hi=0 | tensor_dim1[15:0]
    g1[3] = (((unsigned int)NTOK >> 16) & 0xFFFFu)    // tensor_dim1[31:16]
          | (32u << 16);                              // tile_dim0 = 32
    g1[4] = 64u;                                      // tile_dim1 = 64, tile_dim2 = 0
    g1[5] = (unsigned int)Cbuf;                       // tensor_dim0_stride[31:0]
    g1[6] = 0u;
    g1[7] = 0u;
    asm volatile("tensor_load_to_lds %0, %1" :: "s"(g0), "s"(g1) : "memory");
}

// ---------------------------------------------------------------------------
// Weight repack: OIHW f32 -> [Mpad][Kpad] f16 with k = (ky*ks+kx)*C + c
// ---------------------------------------------------------------------------
__global__ void k_pack_w(const float* __restrict__ src, _Float16* __restrict__ dst,
                         int M, int C, int ks, int Mpad, int Kpad) {
    int i = blockIdx.x * blockDim.x + threadIdx.x;
    if (i >= Mpad * Kpad) return;
    int m = i / Kpad, kk = i - m * Kpad;
    float v = 0.f;
    if (m < M && kk < ks * ks * C) {
        int kpos = kk / C, c = kk - kpos * C;
        int ky = kpos / ks, kx = kpos - ky * ks;
        v = src[(((size_t)m * C + c) * ks + ky) * ks + kx];
    }
    dst[i] = (_Float16)v;
}

// scale/shift precompute: BN -> (inv, b - m*inv)
__global__ void k_bn_prep(const float* g, const float* b, const float* m, const float* v,
                          float* scale, float* shift, int C) {
    int c = blockIdx.x * blockDim.x + threadIdx.x;
    if (c >= C) return;
    float inv = g[c] * rsqrtf(v[c] + EPSV);
    scale[c] = inv;
    shift[c] = b[c] - m[c] * inv;
}

// bias-only epilogue (padded region -> 1,0)
__global__ void k_bias_prep(const float* b, float* scale, float* shift, int C, int Cvalid) {
    int c = blockIdx.x * blockDim.x + threadIdx.x;
    if (c >= C) return;
    scale[c] = 1.f;
    shift[c] = (c < Cvalid) ? b[c] : 0.f;
}

// sap bias fused with bn2:  y = (x + b_sap)*inv2 + (b2 - m2*inv2)
__global__ void k_sapbn2_prep(const float* sapb, const float* g, const float* b,
                              const float* m, const float* v, float* scale, float* shift, int C) {
    int c = blockIdx.x * blockDim.x + threadIdx.x;
    if (c >= C) return;
    float inv = g[c] * rsqrtf(v[c] + EPSV);
    scale[c] = inv;
    shift[c] = sapb[c] * inv + (b[c] - m[c] * inv);
}

// NCHW f32 -> NHWC f16
__global__ void k_to_nhwc(const float* __restrict__ in, _Float16* __restrict__ out, int C) {
    int i = blockIdx.x * blockDim.x + threadIdx.x;
    if (i >= NTOK * C) return;
    int c = i % C;
    int n = i / C;
    int b = n / HWSZ, p = n - b * HWSZ;
    out[i] = (_Float16)in[((size_t)b * C + c) * HWSZ + p];
}

// ---------------------------------------------------------------------------
// Implicit-GEMM conv / linear with WMMA.
//   out[n][m] = act( scale[m]*(sum_k A[m,k]*im2col[n,k]) + shift[m] [+ res] )
// A: f16 [Mtot][Kpad]; activations: NHWC f16 (Cbuf channels). DUP: channel
// wrap for the cat([fm,fm]) pooled input.  Block: 256 thr = 8 waves; each
// wave one 16-row M tile x 64-pixel N tile; 32-deep k steps via LDS.
// USE_TDM (KSZ==1, !DUP): B tiles DMA'd by the Tensor Data Mover, ping-pong.
// ---------------------------------------------------------------------------
template <int KSZ, int ACT, bool DUP, bool HAS_RES, bool OUT16, bool OUT32, bool OUT_NCHW,
          bool USE_TDM>
__global__ __launch_bounds__(256)
void k_conv_gemm(const _Float16* __restrict__ Aw,
                 const float* __restrict__ scale, const float* __restrict__ shift,
                 const _Float16* __restrict__ act, int Cbuf, int Clog,
                 int Kpad, int Mtot,
                 const float* __restrict__ res,
                 _Float16* __restrict__ out16, int ldo16, int coff16,
                 float* __restrict__ out32, int ldo32, int coff32,
                 float* __restrict__ outN) {
    __shared__ _Float16 Bt[2][64][40];              // ping-pong 64 px x 32 k (+pad)
    const int n0    = blockIdx.x * 64;
    const int mbase = blockIdx.y * 128;
    const int tid   = threadIdx.x;
    const int wave  = tid >> 5, lane = tid & 31;
    const int lhalf = lane >> 4, l15 = lane & 15;
    const int mtile = mbase + wave * 16;
    const bool compute = (mtile < Mtot);
    const int PADP = KSZ / 2;
    const int nk = Kpad / 32;

    v8f acc[4] = {};

    if constexpr (USE_TDM) {
        if (wave == 0)
            tdm_load_tile(act + (size_t)n0 * Cbuf,
                          (unsigned int)(uintptr_t)&Bt[0][0][0], Cbuf);
    }

    for (int i = 0; i < nk; ++i) {
        const int k0  = i * 32;
        const int buf = USE_TDM ? (i & 1) : 0;
        if constexpr (USE_TDM) {
            __syncthreads();                         // prev compute done -> safe overwrite
            if (wave == 0) {
                if (i + 1 < nk) {
                    tdm_load_tile(act + (size_t)n0 * Cbuf + (i + 1) * 32,
                                  (unsigned int)(uintptr_t)&Bt[buf ^ 1][0][0], Cbuf);
                    __builtin_amdgcn_s_wait_tensorcnt(1);   // current tile landed
                } else {
                    __builtin_amdgcn_s_wait_tensorcnt(0);
                }
            }
            __syncthreads();                         // publish tile to all waves
        } else {
            // ---- cooperative B tile load (implicit im2col gather) ----
            int pl = tid >> 2;                      // 0..63 pixel in tile
            int kq = (tid & 3) * 8;                 // 0,8,16,24
            int n  = n0 + pl;
            int b  = n / HWSZ, p = n - b * HWSZ;
            int y  = p / WID, x = p - y * WID;
            int kk = k0 + kq;
            v8h val = {};
            if (kk < Clog * KSZ * KSZ) {
                int kpos = kk / Clog;
                int c    = kk - kpos * Clog;
                if (DUP) c &= (Cbuf - 1);
                int ky = kpos / KSZ, kx = kpos - ky * KSZ;
                int sy = y + ky - PADP, sx = x + kx - PADP;
                if (sy >= 0 && sy < HH && sx >= 0 && sx < WID)
                    val = *(const v8h*)(act + (((size_t)b * HWSZ + sy * WID + sx) * Cbuf + c));
            }
            *(v8h*)(&Bt[0][pl][kq]) = val;
            __syncthreads();
        }
        if (compute) {
            // A fragment: lane half picks K in {0..7,16..23} vs {8..15,24..31}
            const _Float16* ap = Aw + (size_t)(mtile + l15) * Kpad + k0 + lhalf * 8;
            __builtin_prefetch(ap + 32, 0, 3);      // next k-step, L2 HT
            v8h a_lo = *(const v8h*)(ap);
            v8h a_hi = *(const v8h*)(ap + 16);
            v16h afrag = SHUF16(a_lo, a_hi);
#pragma unroll
            for (int nt = 0; nt < 4; ++nt) {
                const _Float16* bp = &Bt[buf][nt * 16 + l15][lhalf * 8];
                v8h b_lo = *(const v8h*)(bp);
                v8h b_hi = *(const v8h*)(bp + 16);
                v16h bfrag = SHUF16(b_lo, b_hi);
                acc[nt] = __builtin_amdgcn_wmma_f32_16x16x32_f16(
                    false, afrag, false, bfrag, (short)0, acc[nt], false, false);
            }
        }
        if constexpr (!USE_TDM) __syncthreads();
    }
    if (!compute) return;

#pragma unroll
    for (int nt = 0; nt < 4; ++nt) {
        int n = n0 + nt * 16 + l15;
#pragma unroll
        for (int r = 0; r < 8; ++r) {
            int m = mtile + lhalf * 8 + r;          // D layout: lanes 16..31 -> M+8
            float v = acc[nt][r];
            v = v * scale[m] + shift[m];
            if (HAS_RES) v += res[(size_t)n * Mtot + m];
            if (ACT == 1) v = v > 0.f ? v : 0.f;
            if (ACT == 2) v = gelu_exact(v);
            if (OUT16) out16[(size_t)n * ldo16 + coff16 + m] = (_Float16)v;
            if (OUT32) out32[(size_t)n * ldo32 + coff32 + m] = v;
            if (OUT_NCHW) {
                int b = n / HWSZ, p = n - b * HWSZ;
                outN[((size_t)b * Mtot + m) * HWSZ + p] = v;
            }
        }
    }
}

// ---------------------------------------------------------------------------
// Windowed self-attention: one wave per (batch, 7x7 window, head).
// 49 tokens padded to 64; S = QK^T (16 WMMAs), masked softmax, O = PV (8).
// ---------------------------------------------------------------------------
__global__ __launch_bounds__(32)
void k_win_attn(const _Float16* __restrict__ q, const _Float16* __restrict__ k,
                const _Float16* __restrict__ v, _Float16* __restrict__ o) {
    __shared__ _Float16 qs[64][32];
    __shared__ _Float16 ks[64][32];
    __shared__ _Float16 vT[16][64];
    __shared__ float    sb[64][65];
    __shared__ _Float16 pb[64][64];

    const int head = blockIdx.x & 7;
    const int win  = (blockIdx.x >> 3) & 255;
    const int b    = blockIdx.x >> 11;
    const int wy = win >> 4, wx = win & 15;
    const int lane = threadIdx.x;
    const int lhalf = lane >> 4, l15 = lane & 15;

    // stage Q,K (64x32, zero-padded) and V^T (16x64)
    for (int t = lane; t < 64; t += 32) {
        v8h z = {};
        if (t < 49) {
            int ty = wy * 7 + t / 7, tx = wx * 7 + t % 7;
            size_t base = ((size_t)b * HWSZ + ty * WID + tx) * DIMC + head * HD;
            v8h qlo = *(const v8h*)(q + base), qhi = *(const v8h*)(q + base + 8);
            v8h klo = *(const v8h*)(k + base), khi = *(const v8h*)(k + base + 8);
            v8h vlo = *(const v8h*)(v + base), vhi = *(const v8h*)(v + base + 8);
            *(v8h*)(&qs[t][0]) = qlo; *(v8h*)(&qs[t][8]) = qhi;
            *(v8h*)(&ks[t][0]) = klo; *(v8h*)(&ks[t][8]) = khi;
#pragma unroll
            for (int d = 0; d < 8; ++d) { vT[d][t] = vlo[d]; vT[d + 8][t] = vhi[d]; }
        } else {
            *(v8h*)(&qs[t][0]) = z; *(v8h*)(&qs[t][8]) = z;
            *(v8h*)(&ks[t][0]) = z; *(v8h*)(&ks[t][8]) = z;
#pragma unroll
            for (int d = 0; d < 16; ++d) vT[d][t] = (_Float16)0.f;
        }
        *(v8h*)(&qs[t][16]) = z; *(v8h*)(&qs[t][24]) = z;   // K-dim zero pad
        *(v8h*)(&ks[t][16]) = z; *(v8h*)(&ks[t][24]) = z;
    }
    __syncthreads();

    // S = Q K^T  (K-dim = HD=16 zero-padded to 32)
    v8f s[4][4] = {};
#pragma unroll
    for (int mi = 0; mi < 4; ++mi) {
        const _Float16* ap = &qs[mi * 16 + l15][lhalf * 8];
        v16h af = SHUF16(*(const v8h*)ap, *(const v8h*)(ap + 16));
#pragma unroll
        for (int ni = 0; ni < 4; ++ni) {
            const _Float16* bp = &ks[ni * 16 + l15][lhalf * 8];
            v16h bf = SHUF16(*(const v8h*)bp, *(const v8h*)(bp + 16));
            s[mi][ni] = __builtin_amdgcn_wmma_f32_16x16x32_f16(
                false, af, false, bf, (short)0, s[mi][ni], false, false);
        }
    }
    // scale + mask -> LDS
#pragma unroll
    for (int mi = 0; mi < 4; ++mi)
#pragma unroll
        for (int ni = 0; ni < 4; ++ni) {
            int col = ni * 16 + l15;
#pragma unroll
            for (int r = 0; r < 8; ++r) {
                float val = s[mi][ni][r] * SCL;
                if (col >= 49) val = -1e30f;
                sb[mi * 16 + lhalf * 8 + r][col] = val;
            }
        }
    __syncthreads();

    // softmax rows (49 valid cols), write P as f16
    for (int row = lane; row < 64; row += 32) {
        float mx = -1e30f;
        for (int c = 0; c < 49; ++c) mx = fmaxf(mx, sb[row][c]);
        float sum = 0.f;
        for (int c = 0; c < 49; ++c) sum += __expf(sb[row][c] - mx);
        float inv = 1.f / sum;
        for (int c = 0; c < 64; ++c)
            pb[row][c] = (c < 49) ? (_Float16)(__expf(sb[row][c] - mx) * inv) : (_Float16)0.f;
    }
    __syncthreads();

    // O = P V   (M=64 tokens, N=16 dims, K=64 tokens in 2 steps)
    v8f oacc[4] = {};
#pragma unroll
    for (int mi = 0; mi < 4; ++mi) {
#pragma unroll
        for (int kc = 0; kc < 2; ++kc) {
            const _Float16* ap = &pb[mi * 16 + l15][kc * 32 + lhalf * 8];
            v16h af = SHUF16(*(const v8h*)ap, *(const v8h*)(ap + 16));
            const _Float16* bp = &vT[l15][kc * 32 + lhalf * 8];
            v16h bf = SHUF16(*(const v8h*)bp, *(const v8h*)(bp + 16));
            oacc[mi] = __builtin_amdgcn_wmma_f32_16x16x32_f16(
                false, af, false, bf, (short)0, oacc[mi], false, false);
        }
    }
#pragma unroll
    for (int mi = 0; mi < 4; ++mi)
#pragma unroll
        for (int r = 0; r < 8; ++r) {
            int qt = mi * 16 + lhalf * 8 + r;
            if (qt < 49) {
                int ty = wy * 7 + qt / 7, tx = wx * 7 + qt % 7;
                o[((size_t)b * HWSZ + ty * WID + tx) * DIMC + head * HD + l15] =
                    (_Float16)oacc[mi][r];
            }
        }
}

// ---------------------------------------------------------------------------
// LayerNorm over C=128 (one wave per token)
// ---------------------------------------------------------------------------
__global__ __launch_bounds__(256)
void k_layernorm(const _Float16* __restrict__ in, const float* __restrict__ g,
                 const float* __restrict__ b, _Float16* __restrict__ out) {
    int tok = blockIdx.x * 8 + (threadIdx.x >> 5);
    int lane = threadIdx.x & 31;
    if (tok >= NTOK) return;
    const _Float16* p = in + (size_t)tok * DIMC;
    float vals[4], sum = 0.f, sq = 0.f;
#pragma unroll
    for (int i = 0; i < 4; ++i) {
        vals[i] = (float)p[lane * 4 + i];
        sum += vals[i]; sq += vals[i] * vals[i];
    }
#pragma unroll
    for (int off = 16; off >= 1; off >>= 1) {
        sum += __shfl_xor(sum, off);
        sq  += __shfl_xor(sq, off);
    }
    float mu = sum * (1.f / DIMC);
    float var = sq * (1.f / DIMC) - mu * mu;
    float inv = rsqrtf(var + EPSV);
    _Float16* q = out + (size_t)tok * DIMC;
#pragma unroll
    for (int i = 0; i < 4; ++i) {
        int c = lane * 4 + i;
        q[c] = (_Float16)((vals[i] - mu) * inv * g[c] + b[c]);
    }
}

// ---------------------------------------------------------------------------
// ai head mixing + softmax over q (9):  attn[n][h][p][q]
// ---------------------------------------------------------------------------
__global__ void k_attn_mix(const _Float16* __restrict__ afg, const _Float16* __restrict__ abg,
                           const float* __restrict__ aiw, const float* __restrict__ aib,
                           float* __restrict__ attn) {
    int i = blockIdx.x * blockDim.x + threadIdx.x;
    if (i >= NTOK * HEADS * 9) return;
    int n = i / 72, r = i - n * 72;
    int h = r / 9, p = r - h * 9;
    float acc[9];
#pragma unroll
    for (int q = 0; q < 9; ++q) acc[q] = aib[h];
    const _Float16* fr = afg + (size_t)n * 768;
    const _Float16* br = abg + (size_t)n * 768;
#pragma unroll
    for (int j = 0; j < 8; ++j) {
        float wf = aiw[h * 16 + j] * SCL;
        float wb = aiw[h * 16 + 8 + j] * SCL;
        int base = j * 81 + p * 9;
#pragma unroll
        for (int q = 0; q < 9; ++q)
            acc[q] += wf * (float)fr[base + q] + wb * (float)br[base + q];
    }
    float mx = acc[0];
#pragma unroll
    for (int q = 1; q < 9; ++q) mx = fmaxf(mx, acc[q]);
    float s = 0.f;
#pragma unroll
    for (int q = 0; q < 9; ++q) { acc[q] = __expf(acc[q] - mx); s += acc[q]; }
    float inv = 1.f / s;
    float* o = attn + (size_t)n * 648 + h * 81 + p * 9;
#pragma unroll
    for (int q = 0; q < 9; ++q) o[q] = acc[q] * inv;
}

// ---------------------------------------------------------------------------
// Aggregation: agg[n][p][c] = sum_q attn[n][h][p][q] * vmap[nbr(q)][c]
// One block (128 thr = channels) per token; neighborhood staged in LDS.
// ---------------------------------------------------------------------------
__global__ __launch_bounds__(128)
void k_agg(const float* __restrict__ attn, const _Float16* __restrict__ vmap,
           _Float16* __restrict__ agg) {
    __shared__ float    at[648];
    __shared__ _Float16 vl[9][128];
    int n = blockIdx.x;
    int b = n / HWSZ, p = n - b * HWSZ;
    int y = p / WID, x = p - y * WID;
    int c = threadIdx.x, h = c >> 4;
    for (int i = c; i < 648; i += 128) at[i] = attn[(size_t)n * 648 + i];
#pragma unroll
    for (int q = 0; q < 9; ++q) {
        int sy = y + q / 3 - 1, sx = x + q % 3 - 1;
        vl[q][c] = (sy >= 0 && sy < HH && sx >= 0 && sx < WID)
                       ? vmap[((size_t)b * HWSZ + sy * WID + sx) * DIMC + c]
                       : (_Float16)0.f;
    }
    __syncthreads();
#pragma unroll
    for (int pp = 0; pp < 9; ++pp) {
        float s = 0.f;
#pragma unroll
        for (int q = 0; q < 9; ++q) s += at[h * 81 + pp * 9 + q] * (float)vl[q][c];
        agg[((size_t)n * 9 + pp) * DIMC + c] = (_Float16)s;
    }
}

// fold3 overlap-add (gather form)
__global__ void k_fold(const _Float16* __restrict__ agg, _Float16* __restrict__ out) {
    int i = blockIdx.x * blockDim.x + threadIdx.x;
    if (i >= NTOK * DIMC) return;
    int c = i & 127, n = i >> 7;
    int b = n / HWSZ, p = n - b * HWSZ;
    int y = p / WID, x = p - y * WID;
    float s = 0.f;
#pragma unroll
    for (int ky = 0; ky < 3; ++ky)
#pragma unroll
        for (int kx = 0; kx < 3; ++kx) {
            int ty = y + 1 - ky, tx = x + 1 - kx;
            if (ty >= 0 && ty < HH && tx >= 0 && tx < WID)
                s += (float)agg[(((size_t)b * HWSZ + ty * WID + tx) * 9 + ky * 3 + kx) * DIMC + c];
        }
    out[i] = (_Float16)s;
}

// ---------------------------------------------------------------------------
extern "C" void kernel_launch(void* const* d_in, const int* in_sizes, int n_in,
                              void* d_out, int out_size, void* d_ws, size_t ws_size,
                              hipStream_t stream) {
    const float* x    = (const float*)d_in[0];
    const float* fg   = (const float*)d_in[1];
    const float* bg   = (const float*)d_in[2];
    const float* res1_w = (const float*)d_in[3];
    const float* res1_bn[4] = {(const float*)d_in[4], (const float*)d_in[5], (const float*)d_in[6], (const float*)d_in[7]};
    const float* res2_w = (const float*)d_in[8];
    const float* res2_bn[4] = {(const float*)d_in[9], (const float*)d_in[10], (const float*)d_in[11], (const float*)d_in[12]};
    const float* main1_w = (const float*)d_in[13];
    const float* main1_bn[4] = {(const float*)d_in[14], (const float*)d_in[15], (const float*)d_in[16], (const float*)d_in[17]};
    const float* main2_w = (const float*)d_in[18];
    const float* main2_bn[4] = {(const float*)d_in[19], (const float*)d_in[20], (const float*)d_in[21], (const float*)d_in[22]};
    const float* fuse_w = (const float*)d_in[23];
    const float* fuse_bn[4] = {(const float*)d_in[24], (const float*)d_in[25], (const float*)d_in[26], (const float*)d_in[27]};
    const float* saq_w = (const float*)d_in[28]; const float* saq_b = (const float*)d_in[29];
    const float* sak_w = (const float*)d_in[30]; const float* sak_b = (const float*)d_in[31];
    const float* sav_w = (const float*)d_in[32]; const float* sav_b = (const float*)d_in[33];
    const float* sap_w = (const float*)d_in[34]; const float* sap_b = (const float*)d_in[35];
    const float* bn2[4] = {(const float*)d_in[36], (const float*)d_in[37], (const float*)d_in[38], (const float*)d_in[39]};
    const float* v_w = (const float*)d_in[40]; const float* v_b = (const float*)d_in[41];
    const float* pa_w = (const float*)d_in[42];
    const float* pa_bn[4] = {(const float*)d_in[43], (const float*)d_in[44], (const float*)d_in[45], (const float*)d_in[46]};
    const float* ln_g = (const float*)d_in[47]; const float* ln_b = (const float*)d_in[48];
    const float* afg_w = (const float*)d_in[49]; const float* afg_b = (const float*)d_in[50];
    const float* abg_w = (const float*)d_in[51]; const float* abg_b = (const float*)d_in[52];
    const float* ai_w = (const float*)d_in[53]; const float* ai_b = (const float*)d_in[54];
    const float* proj_w = (const float*)d_in[55]; const float* proj_b = (const float*)d_in[56];
    const float* out1_w = (const float*)d_in[57];
    const float* out1_bn[4] = {(const float*)d_in[58], (const float*)d_in[59], (const float*)d_in[60], (const float*)d_in[61]};
    const float* out2_w = (const float*)d_in[62];
    const float* out2_bn[4] = {(const float*)d_in[63], (const float*)d_in[64], (const float*)d_in[65], (const float*)d_in[66]};
    float* outp = (float*)d_out;

    // ---- workspace bump allocator (~330 MB) ----
    char* ws = (char*)d_ws;
    size_t cur = 0;
    auto alloc = [&](size_t bytes) -> void* {
        void* p = ws + cur;
        cur = (cur + bytes + 255) & ~(size_t)255;
        return p;
    };
    auto h = [&](size_t elems) { return (_Float16*)alloc(elems * 2); };
    auto f = [&](size_t elems) { return (float*)alloc(elems * 4); };

    // weights (f16 packed)
    _Float16* w_res1  = h(64 * 64);
    _Float16* w_res2  = h(128 * 576);
    _Float16* w_main1 = h(128 * 576);
    _Float16* w_main2 = h(128 * 1152);
    _Float16* w_fuse  = h(128 * 256);
    _Float16* w_q = h(128 * 128); _Float16* w_k = h(128 * 128); _Float16* w_v = h(128 * 128);
    _Float16* w_sap = h(128 * 128); _Float16* w_vw = h(128 * 128); _Float16* w_proj = h(128 * 128);
    _Float16* w_pa  = h(128 * 256);
    _Float16* w_afg = h(768 * 128); _Float16* w_abg = h(768 * 128);
    _Float16* w_out1 = h(128 * 1152); _Float16* w_out2 = h(128 * 1152);
    // scale/shift
    float* sc_res1 = f(64);  float* sh_res1 = f(64);
    float* sc_res2 = f(128); float* sh_res2 = f(128);
    float* sc_m1 = f(128);   float* sh_m1 = f(128);
    float* sc_m2 = f(128);   float* sh_m2 = f(128);
    float* sc_fu = f(128);   float* sh_fu = f(128);
    float* sc_q = f(128);    float* sh_q = f(128);
    float* sc_k = f(128);    float* sh_k = f(128);
    float* sc_v = f(128);    float* sh_v = f(128);
    float* sc_sap = f(128);  float* sh_sap = f(128);
    float* sc_vw = f(128);   float* sh_vw = f(128);
    float* sc_pa = f(128);   float* sh_pa = f(128);
    float* sc_afg = f(768);  float* sh_afg = f(768);
    float* sc_abg = f(768);  float* sh_abg = f(768);
    float* sc_pr = f(128);   float* sh_pr = f(128);
    float* sc_o1 = f(128);   float* sh_o1 = f(128);
    float* sc_o2 = f(128);   float* sh_o2 = f(128);
    // activations
    _Float16* x16   = h((size_t)NTOK * 64);
    _Float16* r1    = h((size_t)NTOK * 64);
    float*    xs32  = f((size_t)NTOK * 128);
    _Float16* cat   = h((size_t)NTOK * 256);
    _Float16* m1    = h((size_t)NTOK * 128);
    _Float16* xf    = h((size_t)NTOK * 128);
    _Float16* qf    = h((size_t)NTOK * 128);
    _Float16* kf    = h((size_t)NTOK * 128);
    _Float16* vf    = h((size_t)NTOK * 128);
    _Float16* ao    = h((size_t)NTOK * 128);
    _Float16* x2    = h((size_t)NTOK * 128);
    _Float16* vm    = h((size_t)NTOK * 128);
    _Float16* fg16  = h((size_t)NTOK * 128);
    _Float16* bg16  = h((size_t)NTOK * 128);
    _Float16* pafg  = h((size_t)NTOK * 128);
    _Float16* pabg  = h((size_t)NTOK * 128);
    _Float16* lnfg  = h((size_t)NTOK * 128);
    _Float16* lnbg  = h((size_t)NTOK * 128);
    _Float16* afg16 = h((size_t)NTOK * 768);
    _Float16* abg16 = h((size_t)NTOK * 768);
    float*    at32  = f((size_t)NTOK * 648);
    _Float16* agg16 = h((size_t)NTOK * 9 * 128);
    _Float16* fo16  = h((size_t)NTOK * 128);
    _Float16* pr16  = h((size_t)NTOK * 128);
    _Float16* o116  = h((size_t)NTOK * 128);
    (void)ws_size; (void)n_in; (void)in_sizes; (void)out_size;

    auto pack = [&](const float* src, _Float16* dst, int M, int C, int ks, int Mpad, int Kpad) {
        int n = Mpad * Kpad;
        k_pack_w<<<(n + 255) / 256, 256, 0, stream>>>(src, dst, M, C, ks, Mpad, Kpad);
    };
    pack(res1_w, w_res1, 64, 64, 1, 64, 64);
    pack(res2_w, w_res2, 128, 64, 3, 128, 576);
    pack(main1_w, w_main1, 128, 64, 3, 128, 576);
    pack(main2_w, w_main2, 128, 128, 3, 128, 1152);
    pack(fuse_w, w_fuse, 128, 256, 1, 128, 256);
    pack(saq_w, w_q, 128, 128, 1, 128, 128);
    pack(sak_w, w_k, 128, 128, 1, 128, 128);
    pack(sav_w, w_v, 128, 128, 1, 128, 128);
    pack(sap_w, w_sap, 128, 128, 1, 128, 128);
    pack(v_w, w_vw, 128, 128, 1, 128, 128);
    pack(proj_w, w_proj, 128, 128, 1, 128, 128);
    pack(pa_w, w_pa, 128, 256, 1, 128, 256);
    pack(afg_w, w_afg, 648, 128, 1, 768, 128);
    pack(abg_w, w_abg, 648, 128, 1, 768, 128);
    pack(out1_w, w_out1, 128, 128, 3, 128, 1152);
    pack(out2_w, w_out2, 128, 128, 3, 128, 1152);

    auto bnp = [&](const float* const* bn, float* sc, float* sh, int C) {
        k_bn_prep<<<1, 128, 0, stream>>>(bn[0], bn[1], bn[2], bn[3], sc, sh, C);
    };
    bnp(res1_bn, sc_res1, sh_res1, 64);
    bnp(res2_bn, sc_res2, sh_res2, 128);
    bnp(main1_bn, sc_m1, sh_m1, 128);
    bnp(main2_bn, sc_m2, sh_m2, 128);
    bnp(fuse_bn, sc_fu, sh_fu, 128);
    bnp(pa_bn, sc_pa, sh_pa, 128);
    bnp(out1_bn, sc_o1, sh_o1, 128);
    bnp(out2_bn, sc_o2, sh_o2, 128);
    k_bias_prep<<<1, 128, 0, stream>>>(saq_b, sc_q, sh_q, 128, 128);
    k_bias_prep<<<1, 128, 0, stream>>>(sak_b, sc_k, sh_k, 128, 128);
    k_bias_prep<<<1, 128, 0, stream>>>(sav_b, sc_v, sh_v, 128, 128);
    k_bias_prep<<<1, 128, 0, stream>>>(v_b, sc_vw, sh_vw, 128, 128);
    k_bias_prep<<<1, 128, 0, stream>>>(proj_b, sc_pr, sh_pr, 128, 128);
    k_bias_prep<<<3, 256, 0, stream>>>(afg_b, sc_afg, sh_afg, 768, 648);
    k_bias_prep<<<3, 256, 0, stream>>>(abg_b, sc_abg, sh_abg, 768, 648);
    k_sapbn2_prep<<<1, 128, 0, stream>>>(sap_b, bn2[0], bn2[1], bn2[2], bn2[3], sc_sap, sh_sap, 128);

    k_to_nhwc<<<(NTOK * 64 + 255) / 256, 256, 0, stream>>>(x, x16, 64);
    k_to_nhwc<<<(NTOK * 128 + 255) / 256, 256, 0, stream>>>(fg, fg16, 128);
    k_to_nhwc<<<(NTOK * 128 + 255) / 256, 256, 0, stream>>>(bg, bg16, 128);

    const dim3 g1(NTOK / 64, 1), g6(NTOK / 64, 6);
    // res path: res1 (1x1, TDM) -> res2 (3x3, writes cat[:,128:] f16 + xs f32)
    k_conv_gemm<1,0,false,false,true,false,false,true ><<<g1,256,0,stream>>>(w_res1, sc_res1, sh_res1, x16, 64, 64, 64, 64, nullptr, r1, 64, 0, nullptr, 0, 0, nullptr);
    k_conv_gemm<3,0,false,false,true,true ,false,false><<<g1,256,0,stream>>>(w_res2, sc_res2, sh_res2, r1, 64, 64, 576, 128, nullptr, cat, 256, 128, xs32, 128, 0, nullptr);
    // main path: main1 relu -> main2 (writes cat[:,0:128])
    k_conv_gemm<3,1,false,false,true,false,false,false><<<g1,256,0,stream>>>(w_main1, sc_m1, sh_m1, x16, 64, 64, 576, 128, nullptr, m1, 128, 0, nullptr, 0, 0, nullptr);
    k_conv_gemm<3,0,false,false,true,false,false,false><<<g1,256,0,stream>>>(w_main2, sc_m2, sh_m2, m1, 128, 128, 1152, 128, nullptr, cat, 256, 0, nullptr, 0, 0, nullptr);
    // fuse (1x1 on cat, gelu, TDM)
    k_conv_gemm<1,2,false,false,true,false,false,true ><<<g1,256,0,stream>>>(w_fuse, sc_fu, sh_fu, cat, 256, 256, 256, 128, nullptr, xf, 128, 0, nullptr, 0, 0, nullptr);
    // QKV (TDM)
    k_conv_gemm<1,0,false,false,true,false,false,true ><<<g1,256,0,stream>>>(w_q, sc_q, sh_q, xf, 128, 128, 128, 128, nullptr, qf, 128, 0, nullptr, 0, 0, nullptr);
    k_conv_gemm<1,0,false,false,true,false,false,true ><<<g1,256,0,stream>>>(w_k, sc_k, sh_k, xf, 128, 128, 128, 128, nullptr, kf, 128, 0, nullptr, 0, 0, nullptr);
    k_conv_gemm<1,0,false,false,true,false,false,true ><<<g1,256,0,stream>>>(w_v, sc_v, sh_v, xf, 128, 128, 128, 128, nullptr, vf, 128, 0, nullptr, 0, 0, nullptr);
    // window attention (2 batches * 256 windows * 8 heads)
    k_win_attn<<<NB * 256 * 8, 32, 0, stream>>>(qf, kf, vf, ao);
    // sap projection fused with bn2 -> x2 (TDM)
    k_conv_gemm<1,0,false,false,true,false,false,true ><<<g1,256,0,stream>>>(w_sap, sc_sap, sh_sap, ao, 128, 128, 128, 128, nullptr, x2, 128, 0, nullptr, 0, 0, nullptr);
    // value map (TDM)
    k_conv_gemm<1,0,false,false,true,false,false,true ><<<g1,256,0,stream>>>(w_vw, sc_vw, sh_vw, x2, 128, 128, 128, 128, nullptr, vm, 128, 0, nullptr, 0, 0, nullptr);
    // contrast branches: pa (1x1 on cat([fm,fm]) via DUP) relu -> layernorm -> afg/abg
    k_conv_gemm<1,1,true ,false,true,false,false,false><<<g1,256,0,stream>>>(w_pa, sc_pa, sh_pa, fg16, 128, 256, 256, 128, nullptr, pafg, 128, 0, nullptr, 0, 0, nullptr);
    k_conv_gemm<1,1,true ,false,true,false,false,false><<<g1,256,0,stream>>>(w_pa, sc_pa, sh_pa, bg16, 128, 256, 256, 128, nullptr, pabg, 128, 0, nullptr, 0, 0, nullptr);
    k_layernorm<<<(NTOK + 7) / 8, 256, 0, stream>>>(pafg, ln_g, ln_b, lnfg);
    k_layernorm<<<(NTOK + 7) / 8, 256, 0, stream>>>(pabg, ln_g, ln_b, lnbg);
    k_conv_gemm<1,0,false,false,true,false,false,true ><<<g6,256,0,stream>>>(w_afg, sc_afg, sh_afg, lnfg, 128, 128, 128, 768, nullptr, afg16, 768, 0, nullptr, 0, 0, nullptr);
    k_conv_gemm<1,0,false,false,true,false,false,true ><<<g6,256,0,stream>>>(w_abg, sc_abg, sh_abg, lnbg, 128, 128, 128, 768, nullptr, abg16, 768, 0, nullptr, 0, 0, nullptr);
    // ai mixing + softmax, aggregation, fold
    k_attn_mix<<<(NTOK * 72 + 255) / 256, 256, 0, stream>>>(afg16, abg16, ai_w, ai_b, at32);
    k_agg<<<NTOK, 128, 0, stream>>>(at32, vm, agg16);
    k_fold<<<(NTOK * 128 + 255) / 256, 256, 0, stream>>>(agg16, fo16);
    // proj -> out1 (relu) -> out2 (+xs residual, gelu, NCHW f32 to d_out)
    k_conv_gemm<1,0,false,false,true,false,false,true ><<<g1,256,0,stream>>>(w_proj, sc_pr, sh_pr, fo16, 128, 128, 128, 128, nullptr, pr16, 128, 0, nullptr, 0, 0, nullptr);
    k_conv_gemm<3,1,false,false,true,false,false,false><<<g1,256,0,stream>>>(w_out1, sc_o1, sh_o1, pr16, 128, 128, 1152, 128, nullptr, o116, 128, 0, nullptr, 0, 0, nullptr);
    k_conv_gemm<3,2,false,true ,false,false,true ,false><<<g1,256,0,stream>>>(w_out2, sc_o2, sh_o2, o116, 128, 128, 1152, 128, xs32, nullptr, 0, 0, nullptr, 0, 0, outp);
}
